// DeepAR_82669530513629
// MI455X (gfx1250) — matrix operand order repaired and checked by reading, not compiled, not run on hardware
//
#include <hip/hip_runtime.h>
#include <hip/hip_bf16.h>

#define Bn   4096
#define Sn   512
#define HZn  64
#define Fn   16
#define En   64
#define Hn   256
#define Tn   (Sn + HZn)

#define BT   32      // batch rows per workgroup (2 M-tiles -> 2x B-fragment reuse)
#define K0   352     // layer0 K: [x(16)|emb(64)|pad(16)] + h(256)
#define K1   512     // layer1 K: h_a2(256) + h_b(256)

#define SMEM_BYTES 71808

typedef __attribute__((ext_vector_type(16))) __bf16 bfx16;
typedef __attribute__((ext_vector_type(8)))  __bf16 bfx8;
typedef __attribute__((ext_vector_type(8)))  float  fx8;

// Explicit global (address_space(1)) pointer types so the laundered weight
// base lowers to global_load_b128 (LOADcnt only) instead of flat_load_b128
// (which ties up DScnt + the LDS path needed by the A-fragment ds_loads).
typedef __attribute__((address_space(1))) const __bf16 gbf;
typedef __attribute__((address_space(1))) const bfx16 gvx;

static __device__ __forceinline__ fx8 wmma_bf16(bfx16 a, bfx16 b, fx8 c) {
  // D = A(16x32 bf16) x B(32x16 bf16) + C(16x16 f32)
  return __builtin_amdgcn_wmma_f32_16x16x32_bf16(false, a, false, b, (short)0, c,
                                                 false, false);
}

// A-fragment from LDS (contiguous K order, matching B):
// lanes 0-15: row=lane, K=[kb..kb+15]
// lanes 16-31: row=lane-16, K=[kb+16..kb+31]
static __device__ __forceinline__ bfx16 lds_afrag(const __bf16* base, int rs,
                                                  int kbase, int lane) {
  const int row = lane & 15;
  const int hf  = lane >> 4;
  const __bf16* p = base + row * rs + kbase + hf * 16;
  bfx8 lo = *(const bfx8*)(p);
  bfx8 hi = *(const bfx8*)(p + 8);
  return __builtin_shufflevector(lo, hi, 0, 1, 2, 3, 4, 5, 6, 7,
                                 8, 9, 10, 11, 12, 13, 14, 15);
}

static __device__ __forceinline__ float sigmf(float x) {
  return 1.0f / (1.0f + __expf(-x));
}
static __device__ __forceinline__ float tanh_fast(float x) {
  x = fminf(fmaxf(x, -15.0f), 15.0f);
  const float t = __expf(2.0f * x);
  return (t - 1.0f) / (t + 1.0f);
}
static __device__ __forceinline__ float softplusf(float x) {
  return (x > 20.0f) ? x : log1pf(__expf(x));
}

// ---------------- prep: pack weights to bf16 [N][K] with zero K-padding ----
__global__ void deepar_prep(const float* __restrict__ Wih0, const float* __restrict__ Whh0,
                            const float* __restrict__ bih0, const float* __restrict__ bhh0,
                            const float* __restrict__ Wih1, const float* __restrict__ Whh1,
                            const float* __restrict__ bih1, const float* __restrict__ bhh1,
                            __bf16* __restrict__ W0p, __bf16* __restrict__ W1p,
                            float* __restrict__ b0, float* __restrict__ b1) {
  const int n = blockIdx.x;          // 0..1023 (gate-major rows, torch i,f,g,o)
  const int t = threadIdx.x;
  for (int k = t; k < K0; k += blockDim.x) {
    float v = 0.0f;
    if (k < Fn + En)      v = Wih0[n * (Fn + En) + k];
    else if (k >= 96)     v = Whh0[n * Hn + (k - 96)];
    W0p[(size_t)n * K0 + k] = (__bf16)v;
  }
  for (int k = t; k < K1; k += blockDim.x) {
    float v = (k < Hn) ? Wih1[n * Hn + k] : Whh1[n * Hn + (k - Hn)];
    W1p[(size_t)n * K1 + k] = (__bf16)v;
  }
  if (t == 0) {
    b0[n] = bih0[n] + bhh0[n];
    b1[n] = bih1[n] + bhh1[n];
  }
}

// ---------------- main persistent recurrent kernel -------------------------
__launch_bounds__(256)
__global__ void deepar_main(const float* __restrict__ X, const float* __restrict__ yv,
                            const float* __restrict__ Xf, const float* __restrict__ eps,
                            const float* __restrict__ W_e, const float* __restrict__ b_e,
                            const __bf16* __restrict__ W0p, const __bf16* __restrict__ W1p,
                            const float* __restrict__ bias0, const float* __restrict__ bias1,
                            const float* __restrict__ Wmu, const float* __restrict__ bmu,
                            const float* __restrict__ Wsig, const float* __restrict__ bsig,
                            float* __restrict__ out) {
  extern __shared__ char smem[];
  __bf16* xe    = (__bf16*)(smem);                          // 32 x 96  = 6144 B
  __bf16* hAbuf = (__bf16*)(smem + 6144);                   // 2 x 32 x 256 = 32768 B
  __bf16* hBbuf = (__bf16*)(smem + 6144 + 32768);           // 2 x 32 x 256 = 32768 B
  float*  yprev = (float*)(smem + 6144 + 32768 + 32768);    // 32 x 4 = 128 B

  const int tid  = threadIdx.x;
  const int lane = tid & 31;
  const int wv   = tid >> 5;          // 8 waves
  const int b0r  = blockIdx.x * BT;
  const int colh = lane & 15;         // N within tile (C/D layout)
  const int half = lane >> 4;         // M-half selector

  // zero-init state
  for (int i = tid; i < BT * Hn; i += blockDim.x) {
    hAbuf[i] = (__bf16)0.0f;
    hBbuf[i] = (__bf16)0.0f;
  }
  if (tid < BT) yprev[tid] = 0.0f;
  __syncthreads();

  // cell states live in registers (lane-owned per C/D layout):
  // cX<m-tile><nt-index>
  fx8 cA00 = {}, cA01 = {}, cA10 = {}, cA11 = {};
  fx8 cB00 = {}, cB01 = {}, cB10 = {}, cB11 = {};

  float* out_pred = out;
  float* out_mu   = out + (size_t)Bn * HZn;
  float* out_sig  = out + (size_t)Bn * HZn + (size_t)Bn * Tn;

  for (int t = 0; t < Tn; ++t) {
    const __bf16* hAc = hAbuf + (t & 1) * (BT * Hn);
    __bf16*       hAn = hAbuf + ((t & 1) ^ 1) * (BT * Hn);
    const __bf16* hBc = hBbuf + (t & 1) * (BT * Hn);
    __bf16*       hBn = hBbuf + ((t & 1) ^ 1) * (BT * Hn);

    // Launder weight base pointers each step so loop-invariant code motion
    // cannot hoist the B-fragment loads out of the t-loop (which would cause
    // massive scratch spills). "+s" keeps the base uniform in SGPRs, and the
    // address_space(1) cast keeps the loads on the GLOBAL path.
    unsigned long long w0a = (unsigned long long)W0p;
    unsigned long long w1a = (unsigned long long)W1p;
    asm volatile("" : "+s"(w0a));
    asm volatile("" : "+s"(w1a));
    gbf* W0v = (gbf*)w0a;
    gbf* W1v = (gbf*)w1a;

    // ---- phase 1: build [x_t | y_in*W_e+b_e | 0] in bf16 ----
    {
      const int r  = tid >> 3;        // 32 rows, 8 threads each
      const int c0 = tid & 7;
      const int gb = b0r + r;
      const float y_in = (t < Sn) ? yv[(size_t)gb * Sn + t] : yprev[r];
#pragma unroll
      for (int j = 0; j < 12; ++j) {
        const int c = c0 + j * 8;
        float v;
        if (c < Fn)
          v = (t < Sn) ? X[((size_t)gb * Sn + t) * Fn + c]
                       : Xf[((size_t)gb * HZn + (t - Sn)) * Fn + c];
        else if (c < Fn + En)
          v = y_in * W_e[c - Fn] + b_e[c - Fn];
        else
          v = 0.0f;
        xe[r * 96 + c] = (__bf16)v;
      }
    }
    __syncthreads();

    // ---- phase 2: layer 0 — gates + cell update (2 M-tiles share B) ----
    auto layer0_tile = [&](int nt, fx8& c_m0, fx8& c_m1) {
      const int ncol = nt * 16 + colh;
      gbf* w0 = W0v + (size_t)(0 * Hn + ncol) * K0 + half * 16;
      gbf* w1 = W0v + (size_t)(1 * Hn + ncol) * K0 + half * 16;
      gbf* w2 = W0v + (size_t)(2 * Hn + ncol) * K0 + half * 16;
      gbf* w3 = W0v + (size_t)(3 * Hn + ncol) * K0 + half * 16;
      fx8 aI0 = {}, aF0 = {}, aG0 = {}, aO0 = {};
      fx8 aI1 = {}, aF1 = {}, aG1 = {}, aO1 = {};
#pragma unroll
      for (int kt = 0; kt < 11; ++kt) {
        bfx16 a0, a1;
        if (kt < 3) {
          a0 = lds_afrag(xe, 96, kt * 32, lane);
          a1 = lds_afrag(xe + 16 * 96, 96, kt * 32, lane);
        } else {
          a0 = lds_afrag(hAc, Hn, (kt - 3) * 32, lane);
          a1 = lds_afrag(hAc + 16 * Hn, Hn, (kt - 3) * 32, lane);
        }
        bfx16 bi = *(gvx*)(w0 + kt * 32);
        bfx16 bf = *(gvx*)(w1 + kt * 32);
        bfx16 bg = *(gvx*)(w2 + kt * 32);
        bfx16 bo = *(gvx*)(w3 + kt * 32);
        aI0 = wmma_bf16(a0, bi, aI0);  aI1 = wmma_bf16(a1, bi, aI1);
        aF0 = wmma_bf16(a0, bf, aF0);  aF1 = wmma_bf16(a1, bf, aF1);
        aG0 = wmma_bf16(a0, bg, aG0);  aG1 = wmma_bf16(a1, bg, aG1);
        aO0 = wmma_bf16(a0, bo, aO0);  aO1 = wmma_bf16(a1, bo, aO1);
      }
      const float bI = bias0[0 * Hn + ncol], bF = bias0[1 * Hn + ncol];
      const float bG = bias0[2 * Hn + ncol], bO = bias0[3 * Hn + ncol];
#pragma unroll
      for (int v = 0; v < 8; ++v) {
        const int m = v + half * 8;
        {
          const float c2 = sigmf(aF0[v] + bF) * c_m0[v] +
                           sigmf(aI0[v] + bI) * tanh_fast(aG0[v] + bG);
          c_m0[v] = c2;
          hAn[m * Hn + ncol] = (__bf16)(sigmf(aO0[v] + bO) * tanh_fast(c2));
        }
        {
          const float c2 = sigmf(aF1[v] + bF) * c_m1[v] +
                           sigmf(aI1[v] + bI) * tanh_fast(aG1[v] + bG);
          c_m1[v] = c2;
          hAn[(16 + m) * Hn + ncol] = (__bf16)(sigmf(aO1[v] + bO) * tanh_fast(c2));
        }
      }
    };
    layer0_tile(wv, cA00, cA10);
    layer0_tile(wv + 8, cA01, cA11);
    __syncthreads();

    // ---- phase 3: layer 1 — A = [h_a2 | h_b_old] ----
    auto layer1_tile = [&](int nt, fx8& c_m0, fx8& c_m1) {
      const int ncol = nt * 16 + colh;
      gbf* w0 = W1v + (size_t)(0 * Hn + ncol) * K1 + half * 16;
      gbf* w1 = W1v + (size_t)(1 * Hn + ncol) * K1 + half * 16;
      gbf* w2 = W1v + (size_t)(2 * Hn + ncol) * K1 + half * 16;
      gbf* w3 = W1v + (size_t)(3 * Hn + ncol) * K1 + half * 16;
      fx8 aI0 = {}, aF0 = {}, aG0 = {}, aO0 = {};
      fx8 aI1 = {}, aF1 = {}, aG1 = {}, aO1 = {};
#pragma unroll
      for (int kt = 0; kt < 16; ++kt) {
        bfx16 a0, a1;
        if (kt < 8) {
          a0 = lds_afrag(hAn, Hn, kt * 32, lane);
          a1 = lds_afrag(hAn + 16 * Hn, Hn, kt * 32, lane);
        } else {
          a0 = lds_afrag(hBc, Hn, (kt - 8) * 32, lane);
          a1 = lds_afrag(hBc + 16 * Hn, Hn, (kt - 8) * 32, lane);
        }
        bfx16 bi = *(gvx*)(w0 + kt * 32);
        bfx16 bf = *(gvx*)(w1 + kt * 32);
        bfx16 bg = *(gvx*)(w2 + kt * 32);
        bfx16 bo = *(gvx*)(w3 + kt * 32);
        aI0 = wmma_bf16(a0, bi, aI0);  aI1 = wmma_bf16(a1, bi, aI1);
        aF0 = wmma_bf16(a0, bf, aF0);  aF1 = wmma_bf16(a1, bf, aF1);
        aG0 = wmma_bf16(a0, bg, aG0);  aG1 = wmma_bf16(a1, bg, aG1);
        aO0 = wmma_bf16(a0, bo, aO0);  aO1 = wmma_bf16(a1, bo, aO1);
      }
      const float bI = bias1[0 * Hn + ncol], bF = bias1[1 * Hn + ncol];
      const float bG = bias1[2 * Hn + ncol], bO = bias1[3 * Hn + ncol];
#pragma unroll
      for (int v = 0; v < 8; ++v) {
        const int m = v + half * 8;
        {
          const float c2 = sigmf(aF0[v] + bF) * c_m0[v] +
                           sigmf(aI0[v] + bI) * tanh_fast(aG0[v] + bG);
          c_m0[v] = c2;
          hBn[m * Hn + ncol] = (__bf16)(sigmf(aO0[v] + bO) * tanh_fast(c2));
        }
        {
          const float c2 = sigmf(aF1[v] + bF) * c_m1[v] +
                           sigmf(aI1[v] + bI) * tanh_fast(aG1[v] + bG);
          c_m1[v] = c2;
          hBn[(16 + m) * Hn + ncol] = (__bf16)(sigmf(aO1[v] + bO) * tanh_fast(c2));
        }
      }
    };
    layer1_tile(wv, cB00, cB10);
    layer1_tile(wv + 8, cB01, cB11);
    __syncthreads();

    // ---- phase 4: Gaussian head (relu on the fly), sample, feedback ----
    {
      const int r   = tid >> 3;       // 32 rows, 8 lanes each
      const int seg = tid & 7;
      const int gb  = b0r + r;
      float pm = 0.0f, ps = 0.0f;
#pragma unroll
      for (int j = 0; j < 32; ++j) {
        const int n = seg * 32 + j;
        float hv = (float)hBn[r * Hn + n];
        hv = fmaxf(hv, 0.0f);
        pm += hv * Wmu[n];
        ps += hv * Wsig[n];
      }
#pragma unroll
      for (int msk = 4; msk >= 1; msk >>= 1) {
        pm += __shfl_xor(pm, msk, 32);
        ps += __shfl_xor(ps, msk, 32);
      }
      if (seg == 0) {
        const float mu  = pm + bmu[0];
        const float sig = softplusf(ps + bsig[0]);
        const float ys  = mu + sig * eps[(size_t)t * Bn + gb];
        out_mu[(size_t)gb * Tn + t]  = mu;
        out_sig[(size_t)gb * Tn + t] = sig;
        if (t >= Sn - 1 && t < Sn - 1 + HZn)
          out_pred[(size_t)gb * HZn + (t - (Sn - 1))] = ys;
        yprev[r] = ys;
      }
    }
    __syncthreads();
  }
}

extern "C" void kernel_launch(void* const* d_in, const int* in_sizes, int n_in,
                              void* d_out, int out_size, void* d_ws, size_t ws_size,
                              hipStream_t stream) {
  (void)in_sizes; (void)n_in; (void)out_size; (void)ws_size;
  const float* X    = (const float*)d_in[0];
  const float* yv   = (const float*)d_in[1];
  const float* Xf   = (const float*)d_in[2];
  const float* eps  = (const float*)d_in[3];
  const float* W_e  = (const float*)d_in[4];
  const float* b_e  = (const float*)d_in[5];
  const float* Wih0 = (const float*)d_in[6];
  const float* Whh0 = (const float*)d_in[7];
  const float* bih0 = (const float*)d_in[8];
  const float* bhh0 = (const float*)d_in[9];
  const float* Wih1 = (const float*)d_in[10];
  const float* Whh1 = (const float*)d_in[11];
  const float* bih1 = (const float*)d_in[12];
  const float* bhh1 = (const float*)d_in[13];
  const float* Wmu  = (const float*)d_in[14];
  const float* bmu  = (const float*)d_in[15];
  const float* Wsig = (const float*)d_in[16];
  const float* bsig = (const float*)d_in[17];

  char* ws = (char*)d_ws;
  __bf16* W0p = (__bf16*)ws;                                   // 1024*352*2 = 720896 B
  __bf16* W1p = (__bf16*)(ws + 720896);                        // 1024*512*2 = 1048576 B
  float*  b0  = (float*)(ws + 720896 + 1048576);               // 4096 B
  float*  b1  = (float*)(ws + 720896 + 1048576 + 4096);        // 4096 B

  deepar_prep<<<4 * Hn, 128, 0, stream>>>(Wih0, Whh0, bih0, bhh0,
                                          Wih1, Whh1, bih1, bhh1,
                                          W0p, W1p, b0, b1);

  hipFuncSetAttribute((const void*)deepar_main,
                      hipFuncAttributeMaxDynamicSharedMemorySize, SMEM_BYTES);

  deepar_main<<<Bn / BT, 256, SMEM_BYTES, stream>>>(X, yv, Xf, eps, W_e, b_e,
                                                    W0p, W1p, b0, b1,
                                                    Wmu, bmu, Wsig, bsig,
                                                    (float*)d_out);
}
